// TestModel_8_4355096838981
// MI455X (gfx1250) — compile-verified, matrix operand order checked by reading, not run on hardware
//
#include <hip/hip_runtime.h>
#include <hip/hip_bf16.h>
#include <math.h>

#define DEVI __device__ __forceinline__

typedef __attribute__((ext_vector_type(16))) __bf16 v16bf;
typedef __attribute__((ext_vector_type(8)))  __bf16 v8bf;
typedef __attribute__((ext_vector_type(8)))  float  v8f;

struct alignas(16) U4 { unsigned int x, y, z, w; };
struct alignas(16) F4 { float x, y, z, w; };

constexpr int LAY   = 2;
constexpr int HEADS = 16;
constexpr int AG    = 16;
constexpr int CH    = 512;
constexpr int INTR  = 2048;
constexpr int HDm   = CH / HEADS;     // 32
constexpr int BATCH = 8;
constexpr int TOK   = 2048;
constexpr int MROWS = BATCH * TOK;    // 16384
#define SCALE_F 0.1767766952966369f   // 32^-0.5

DEVI unsigned int f2bf_u(float f) {
  unsigned int u = __float_as_uint(f);
  return (u + 0x7FFFu + ((u >> 16) & 1u)) >> 16;   // RNE f32->bf16
}
DEVI unsigned int pack2(float a, float b) {
  return (f2bf_u(a) & 0xFFFFu) | (f2bf_u(b) << 16);
}

// ---------------------------------------------------------------------------
// WMMA GEMM:  C[M,N] = act[M,K](f32) * Wt[N,K](bf16)  + bias (+resid) (+gelu)
// Block tile 128x128, BK=32, 256 threads = 8 wave32 (4 M x 2 N),
// each wave: 2x4 tiles of 16x16 -> 8 v_wmma_f32_16x16x32_bf16 per K step.
// ---------------------------------------------------------------------------
#define BM 128
#define BN 128
#define BK 32
#define LROW 40   // halves per LDS row (32 data + 8 pad) -> conflict-free b128

__global__ __launch_bounds__(256) void gemm_bf16_wmma(
    const float* __restrict__ Aact, const unsigned short* __restrict__ Bt,
    const float* __restrict__ bias, const float* __restrict__ resid,
    float* __restrict__ Cout, int Mg, int Ng, int Kg, int apply_gelu)
{
  __shared__ unsigned short As[BM * LROW];
  __shared__ unsigned short Bs[BN * LROW];
  const int tid  = threadIdx.x;
  const int lane = tid & 31;
  const int wave = tid >> 5;
  const int wm   = wave & 3;          // wave M index 0..3
  const int wn   = wave >> 2;         // wave N index 0..1
  const int hi   = lane >> 4;         // lane-half selects K sub-block
  const int lr   = lane & 15;
  const int m0   = blockIdx.y * BM;
  const int n0   = blockIdx.x * BN;
  const int ldr  = tid >> 1;          // 0..127 row of tile being staged
  const int ldc  = (tid & 1) * 16;    // half-offset in K chunk

  v8f zero = {};
  v8f acc[2][4];
#pragma unroll
  for (int i = 0; i < 2; ++i)
#pragma unroll
    for (int j = 0; j < 4; ++j) acc[i][j] = zero;

  const float*          ap = Aact + (long)(m0 + ldr) * Kg + ldc;
  const unsigned short* bp = Bt   + (long)(n0 + ldr) * Kg + ldc;

  for (int kk = 0; kk < Kg; kk += BK) {
    F4 a0 = *(const F4*)(ap);
    F4 a1 = *(const F4*)(ap + 4);
    F4 a2 = *(const F4*)(ap + 8);
    F4 a3 = *(const F4*)(ap + 12);
    U4 b0 = *(const U4*)(bp);
    U4 b1 = *(const U4*)(bp + 8);
    if (kk + BK < Kg) {               // global_prefetch_b8 of next tiles
      __builtin_prefetch(ap + BK, 0, 1);
      __builtin_prefetch(bp + BK, 0, 1);
    }
    ap += BK; bp += BK;

    __syncthreads();                  // previous iter's LDS reads done
    U4 ua, ub;
    ua.x = pack2(a0.x, a0.y); ua.y = pack2(a0.z, a0.w);
    ua.z = pack2(a1.x, a1.y); ua.w = pack2(a1.z, a1.w);
    ub.x = pack2(a2.x, a2.y); ub.y = pack2(a2.z, a2.w);
    ub.z = pack2(a3.x, a3.y); ub.w = pack2(a3.z, a3.w);
    *(U4*)&As[ldr * LROW + ldc]     = ua;
    *(U4*)&As[ldr * LROW + ldc + 8] = ub;
    *(U4*)&Bs[ldr * LROW + ldc]     = b0;
    *(U4*)&Bs[ldr * LROW + ldc + 8] = b1;
    __syncthreads();

    // Fragment loads per CDNA5 16-bit A layout:
    // lanes 0-15 hold K{0..7,16..23}, lanes 16-31 hold K{8..15,24..31}.
    v16bf aF[2], bF[4];
#pragma unroll
    for (int tm = 0; tm < 2; ++tm) {
      int r = wm * 32 + tm * 16 + lr;
      v8bf lo  = *(const v8bf*)&As[r * LROW + hi * 8];
      v8bf hi8 = *(const v8bf*)&As[r * LROW + hi * 8 + 16];
      aF[tm] = __builtin_shufflevector(lo, hi8,
                 0,1,2,3,4,5,6,7,8,9,10,11,12,13,14,15);
    }
#pragma unroll
    for (int tn = 0; tn < 4; ++tn) {
      int r = wn * 64 + tn * 16 + lr;
      v8bf lo  = *(const v8bf*)&Bs[r * LROW + hi * 8];
      v8bf hi8 = *(const v8bf*)&Bs[r * LROW + hi * 8 + 16];
      bF[tn] = __builtin_shufflevector(lo, hi8,
                 0,1,2,3,4,5,6,7,8,9,10,11,12,13,14,15);
    }
#pragma unroll
    for (int tm = 0; tm < 2; ++tm)
#pragma unroll
      for (int tn = 0; tn < 4; ++tn)
        acc[tm][tn] = __builtin_amdgcn_wmma_f32_16x16x32_bf16(
            false, aF[tm], false, bF[tn], (short)0, acc[tm][tn], false, false);
  }

  // Epilogue: C layout VGPR r, lane l -> M = r + (l/16)*8, N = l%16
#pragma unroll
  for (int tm = 0; tm < 2; ++tm) {
#pragma unroll
    for (int tn = 0; tn < 4; ++tn) {
      int col = n0 + wn * 64 + tn * 16 + lr;
      float bc = bias ? bias[col] : 0.f;
#pragma unroll
      for (int r = 0; r < 8; ++r) {
        int row = m0 + wm * 32 + tm * 16 + hi * 8 + r;
        long idx = (long)row * Ng + col;
        float x = acc[tm][tn][r] + bc;
        if (resid) x += resid[idx];
        if (apply_gelu) x = 0.5f * x * (1.f + erff(x * 0.7071067811865475f));
        Cout[idx] = x;
      }
    }
  }
}

// ---------------------------------------------------------------------------
// Weight convert: W[K,N] f32  ->  Wt[N,K] bf16 (transposed for B operand)
// ---------------------------------------------------------------------------
__global__ void convw_kernel(const float* __restrict__ W,
                             unsigned short* __restrict__ Wt, int Kk, int Nn)
{
  long idx = (long)blockIdx.x * 256 + threadIdx.x;
  if (idx >= (long)Kk * Nn) return;
  int n = (int)(idx / Kk);
  int k = (int)(idx % Kk);
  Wt[idx] = (unsigned short)f2bf_u(W[(long)k * Nn + n]);
}

// ---------------------------------------------------------------------------
// Relative-position bias precompute: pb[l,h,a], ab[l,h,a]
// 7x7 -> 16x16 half-pixel bilinear then mean, plus axis-bias means.
// ---------------------------------------------------------------------------
__global__ void bias_kernel(const float* __restrict__ an, const float* __restrict__ na,
                            const float* __restrict__ ah, const float* __restrict__ aw,
                            const float* __restrict__ ha, const float* __restrict__ wa,
                            float* __restrict__ pb, float* __restrict__ ab)
{
  int idx = blockIdx.x * 256 + threadIdx.x;          // L*H*A = 512
  if (idx >= LAY * HEADS * AG) return;
  int a = idx % AG;
  int h = (idx / AG) % HEADS;
  int l = idx / (AG * HEADS);
  const float* s1 = an + ((long)(l * HEADS + h) * AG + a) * 49;
  const float* s2 = na + ((long)(l * HEADS + h) * AG + a) * 49;
  float acc1 = 0.f, acc2 = 0.f;
  for (int oy = 0; oy < 16; ++oy) {
    float cy = (oy + 0.5f) * (7.f / 16.f) - 0.5f;
    float fy = floorf(cy);
    float ty = cy - fy;
    int y0 = min(6, max(0, (int)fy));
    int y1 = min(6, max(0, (int)fy + 1));
    for (int ox = 0; ox < 16; ++ox) {
      float cx = (ox + 0.5f) * (7.f / 16.f) - 0.5f;
      float fx = floorf(cx);
      float tx = cx - fx;
      int x0 = min(6, max(0, (int)fx));
      int x1 = min(6, max(0, (int)fx + 1));
      float w00 = (1.f - ty) * (1.f - tx), w01 = (1.f - ty) * tx;
      float w10 = ty * (1.f - tx),         w11 = ty * tx;
      acc1 += w00 * s1[y0 * 7 + x0] + w01 * s1[y0 * 7 + x1]
            + w10 * s1[y1 * 7 + x0] + w11 * s1[y1 * 7 + x1];
      acc2 += w00 * s2[y0 * 7 + x0] + w01 * s2[y0 * 7 + x1]
            + w10 * s2[y1 * 7 + x0] + w11 * s2[y1 * 7 + x1];
    }
  }
  acc1 *= (1.f / 256.f);
  acc2 *= (1.f / 256.f);
  long baseA = ((long)(l * HEADS + h) * AG + a) * AG;  // ah[l,h,a,i] / aw[l,h,a,j]
  long baseH = (long)(l * HEADS + h) * AG * AG;        // ha[l,h,i,0,a] / wa[l,h,0,j,a]
  float sa = 0.f, sw = 0.f, sh = 0.f, sw2 = 0.f;
  for (int i = 0; i < AG; ++i) {
    sa  += ah[baseA + i];
    sw  += aw[baseA + i];
    sh  += ha[baseH + (long)i * AG + a];
    sw2 += wa[baseH + (long)i * AG + a];
  }
  pb[idx] = acc1 + (sa + sw) * (1.f / 16.f);
  ab[idx] = acc2 + (sh + sw2) * (1.f / 16.f);
}

// ---------------------------------------------------------------------------
// Agent pooling: agent[b,a,c] = mean over 128 tokens of q
// ---------------------------------------------------------------------------
__global__ void pool_kernel(const float* __restrict__ q, float* __restrict__ agent)
{
  int idx = blockIdx.x * 256 + threadIdx.x;          // B*A*C = 65536
  if (idx >= BATCH * AG * CH) return;
  int c = idx % CH;
  int a = (idx / CH) % AG;
  int b = idx / (CH * AG);
  const float* qp = q + ((long)b * TOK + (long)a * (TOK / AG)) * CH + c;
  float s = 0.f;
  for (int j = 0; j < TOK / AG; ++j) s += qp[(long)j * CH];
  agent[idx] = s * (1.f / (float)(TOK / AG));
}

// ---------------------------------------------------------------------------
// Stage 1: per (b,h): softmax_n( ag*SCALE . k^T + pb ) @ v -> agent_v[b,h,a,d]
// Wave per agent row; lane=d; online softmax with shfl reductions.
// ---------------------------------------------------------------------------
__global__ __launch_bounds__(256) void stage1_kernel(
    const float* __restrict__ agent, const float* __restrict__ kbuf,
    const float* __restrict__ vbuf,  const float* __restrict__ pb,
    float* __restrict__ agv)
{
  int bh = blockIdx.x;
  int b = bh >> 4, h = bh & 15;
  __shared__ float ag_s[AG][HDm];
  int tid = threadIdx.x;
  for (int i = tid; i < AG * HDm; i += 256) {
    int a = i >> 5, d = i & 31;
    ag_s[a][d] = agent[((long)b * AG + a) * CH + h * HDm + d] * SCALE_F;
  }
  __syncthreads();
  int wave = tid >> 5, lane = tid & 31;
  for (int a = wave; a < AG; a += 8) {
    float bias = pb[h * AG + a];
    float m = -1e30f, sum = 0.f, accd = 0.f;
    const float* kp = kbuf + (long)b * TOK * CH + h * HDm + lane;
    const float* vp = vbuf + (long)b * TOK * CH + h * HDm + lane;
    float agl = ag_s[a][lane];
    for (int n = 0; n < TOK; ++n) {
      float part = agl * kp[(long)n * CH];
#pragma unroll
      for (int off = 16; off; off >>= 1) part += __shfl_xor(part, off, 32);
      float s = part + bias;
      float mnew = fmaxf(m, s);
      float f = __expf(m - mnew);
      float e = __expf(s - mnew);
      accd = accd * f + e * vp[(long)n * CH];
      sum  = sum  * f + e;
      m = mnew;
    }
    agv[(((long)b * HEADS + h) * AG + a) * HDm + lane] = accd / sum;
  }
}

// ---------------------------------------------------------------------------
// Stage 2: per token: softmax_a( q*SCALE . ag^T + ab ) @ agent_v -> ctx[b,n,c]
// ---------------------------------------------------------------------------
__global__ __launch_bounds__(256) void stage2_kernel(
    const float* __restrict__ q, const float* __restrict__ agent,
    const float* __restrict__ agv, const float* __restrict__ ab,
    float* __restrict__ ctx)
{
  int bh = blockIdx.y;
  int b = bh >> 4, h = bh & 15;
  int n = blockIdx.x * 256 + threadIdx.x;
  __shared__ float ag_s[AG][HDm];
  __shared__ float av_s[AG][HDm];
  for (int i = threadIdx.x; i < AG * HDm; i += 256) {
    int a = i >> 5, d = i & 31;
    ag_s[a][d] = agent[((long)b * AG + a) * CH + h * HDm + d];
    av_s[a][d] = agv[(((long)b * HEADS + h) * AG + a) * HDm + d];
  }
  __syncthreads();
  const float* qp = q + ((long)b * TOK + n) * CH + h * HDm;
  float qr[HDm];
#pragma unroll
  for (int d = 0; d < HDm; ++d) qr[d] = qp[d] * SCALE_F;
  float s[AG];
  float mx = -1e30f;
#pragma unroll
  for (int a = 0; a < AG; ++a) {
    float t = ab[h * AG + a];
#pragma unroll
    for (int d = 0; d < HDm; ++d) t += qr[d] * ag_s[a][d];
    s[a] = t;
    mx = fmaxf(mx, t);
  }
  float sum = 0.f;
#pragma unroll
  for (int a = 0; a < AG; ++a) { s[a] = __expf(s[a] - mx); sum += s[a]; }
  float inv = 1.f / sum;
  float* op = ctx + ((long)b * TOK + n) * CH + h * HDm;
#pragma unroll
  for (int d = 0; d < HDm; ++d) {
    float x = 0.f;
#pragma unroll
    for (int a = 0; a < AG; ++a) x += s[a] * av_s[a][d];
    op[d] = x * inv;
  }
}

// ---------------------------------------------------------------------------
// LayerNorm over rows of 512: wave per row.
// ---------------------------------------------------------------------------
__global__ __launch_bounds__(256) void ln_kernel(
    const float* __restrict__ x, const float* __restrict__ g,
    const float* __restrict__ bta, float* __restrict__ y)
{
  int row  = blockIdx.x * 8 + (threadIdx.x >> 5);
  int lane = threadIdx.x & 31;
  const float* xp = x + (long)row * CH;
  float vals[CH / 32];
  float s = 0.f;
#pragma unroll
  for (int i = 0; i < CH / 32; ++i) { vals[i] = xp[lane + i * 32]; s += vals[i]; }
#pragma unroll
  for (int off = 16; off; off >>= 1) s += __shfl_xor(s, off, 32);
  float mu = s * (1.f / CH);
  float v = 0.f;
#pragma unroll
  for (int i = 0; i < CH / 32; ++i) { float d = vals[i] - mu; v += d * d; }
#pragma unroll
  for (int off = 16; off; off >>= 1) v += __shfl_xor(v, off, 32);
  float inv = rsqrtf(v * (1.f / CH) + 1e-12f);
  float* yp = y + (long)row * CH;
#pragma unroll
  for (int i = 0; i < CH / 32; ++i) {
    int c = lane + i * 32;
    yp[c] = (vals[i] - mu) * inv * g[c] + bta[c];
  }
}

// ---------------------------------------------------------------------------
extern "C" void kernel_launch(void* const* d_in, const int* in_sizes, int n_in,
                              void* d_out, int out_size, void* d_ws, size_t ws_size,
                              hipStream_t stream)
{
  (void)in_sizes; (void)n_in; (void)out_size; (void)ws_size;
  const float* s1    = (const float*)d_in[0];
  const float* s2    = (const float*)d_in[1];
  const float* Wq    = (const float*)d_in[2];
  const float* bq    = (const float*)d_in[3];
  const float* Wk    = (const float*)d_in[4];
  const float* bk    = (const float*)d_in[5];
  const float* Wv    = (const float*)d_in[6];
  const float* bv    = (const float*)d_in[7];
  const float* an    = (const float*)d_in[8];
  const float* na    = (const float*)d_in[9];
  const float* ah    = (const float*)d_in[10];
  const float* aw    = (const float*)d_in[11];
  const float* ha    = (const float*)d_in[12];
  const float* wa    = (const float*)d_in[13];
  const float* attnW = (const float*)d_in[14];
  const float* attnb = (const float*)d_in[15];
  const float* ln1g  = (const float*)d_in[16];
  const float* ln1b  = (const float*)d_in[17];
  const float* interW= (const float*)d_in[18];
  const float* interb= (const float*)d_in[19];
  const float* outW  = (const float*)d_in[20];
  const float* outb  = (const float*)d_in[21];
  const float* ln2g  = (const float*)d_in[22];
  const float* ln2b  = (const float*)d_in[23];

  char* ws = (char*)d_ws;
  size_t off = 0;
  auto alloc = [&](size_t bytes) -> char* {
    char* p = ws + off;
    off = (off + bytes + 255) & ~size_t(255);
    return p;
  };

  unsigned short* wt_q[LAY];  unsigned short* wt_k[LAY];  unsigned short* wt_v[LAY];
  unsigned short* wt_at[LAY]; unsigned short* wt_in[LAY]; unsigned short* wt_ou[LAY];
  for (int l = 0; l < LAY; ++l) {
    wt_q[l]  = (unsigned short*)alloc((size_t)CH * CH * 2);
    wt_k[l]  = (unsigned short*)alloc((size_t)CH * CH * 2);
    wt_v[l]  = (unsigned short*)alloc((size_t)CH * CH * 2);
    wt_at[l] = (unsigned short*)alloc((size_t)CH * CH * 2);
    wt_in[l] = (unsigned short*)alloc((size_t)CH * INTR * 2);
    wt_ou[l] = (unsigned short*)alloc((size_t)INTR * CH * 2);
  }
  float* pbb   = (float*)alloc((size_t)LAY * HEADS * AG * 4);
  float* abb   = (float*)alloc((size_t)LAY * HEADS * AG * 4);
  float* agent = (float*)alloc((size_t)BATCH * AG * CH * 4);
  float* agv   = (float*)alloc((size_t)BATCH * HEADS * AG * HDm * 4);
  float* qb    = (float*)alloc((size_t)MROWS * CH * 4);    // q
  float* buf1  = (float*)alloc((size_t)MROWS * CH * 4);    // k -> ctx -> preLN2
  float* buf2  = (float*)alloc((size_t)MROWS * CH * 4);    // v -> preLN1
  float* attn  = (float*)alloc((size_t)MROWS * CH * 4);    // post-LN1
  float* inter = (float*)alloc((size_t)MROWS * INTR * 4);  // FFN hidden
  float* hbuf  = (float*)alloc((size_t)MROWS * CH * 4);    // inter-layer h

  // --- one-time precompute: weights -> bf16 [N][K], rel-pos biases ---
  const int cc  = CH * CH;
  const int ci  = CH * INTR;
  int gcc = (cc + 255) / 256, gci = (ci + 255) / 256;
  for (int l = 0; l < LAY; ++l) {
    convw_kernel<<<gcc, 256, 0, stream>>>(Wq + (size_t)l * cc, wt_q[l], CH, CH);
    convw_kernel<<<gcc, 256, 0, stream>>>(Wk + (size_t)l * cc, wt_k[l], CH, CH);
    convw_kernel<<<gcc, 256, 0, stream>>>(Wv + (size_t)l * cc, wt_v[l], CH, CH);
    convw_kernel<<<gcc, 256, 0, stream>>>(attnW + (size_t)l * cc, wt_at[l], CH, CH);
    convw_kernel<<<gci, 256, 0, stream>>>(interW + (size_t)l * ci, wt_in[l], CH, INTR);
    convw_kernel<<<gci, 256, 0, stream>>>(outW + (size_t)l * ci, wt_ou[l], INTR, CH);
  }
  bias_kernel<<<(LAY * HEADS * AG + 255) / 256, 256, 0, stream>>>(
      an, na, ah, aw, ha, wa, pbb, abb);

  dim3 gC(CH / BN, MROWS / BM);     // N=512 GEMMs
  dim3 gI(INTR / BN, MROWS / BM);   // N=2048 GEMM

  const float* hin = s1;
  for (int l = 0; l < LAY; ++l) {
    // QKV
    gemm_bf16_wmma<<<gC, 256, 0, stream>>>(hin, wt_q[l], bq + (size_t)l * CH,
                                           nullptr, qb, MROWS, CH, CH, 0);
    gemm_bf16_wmma<<<gC, 256, 0, stream>>>(s2, wt_k[l], bk + (size_t)l * CH,
                                           nullptr, buf1, MROWS, CH, CH, 0);
    gemm_bf16_wmma<<<gC, 256, 0, stream>>>(s2, wt_v[l], bv + (size_t)l * CH,
                                           nullptr, buf2, MROWS, CH, CH, 0);
    // agent attention
    pool_kernel<<<(BATCH * AG * CH) / 256, 256, 0, stream>>>(qb, agent);
    stage1_kernel<<<BATCH * HEADS, 256, 0, stream>>>(
        agent, buf1, buf2, pbb + (size_t)l * HEADS * AG, agv);
    stage2_kernel<<<dim3(TOK / 256, BATCH * HEADS), 256, 0, stream>>>(
        qb, agent, agv, abb + (size_t)l * HEADS * AG, buf1 /*ctx*/);
    // attn projection + residual, LN1
    gemm_bf16_wmma<<<gC, 256, 0, stream>>>(buf1, wt_at[l], attnb + (size_t)l * CH,
                                           hin, buf2, MROWS, CH, CH, 0);
    ln_kernel<<<MROWS / 8, 256, 0, stream>>>(buf2, ln1g + (size_t)l * CH,
                                             ln1b + (size_t)l * CH, attn);
    // FFN
    gemm_bf16_wmma<<<gI, 256, 0, stream>>>(attn, wt_in[l], interb + (size_t)l * INTR,
                                           nullptr, inter, MROWS, INTR, CH, 1);
    gemm_bf16_wmma<<<gC, 256, 0, stream>>>(inter, wt_ou[l], outb + (size_t)l * CH,
                                           attn, buf1, MROWS, CH, INTR, 0);
    float* hout = (l == LAY - 1) ? (float*)d_out : hbuf;
    ln_kernel<<<MROWS / 8, 256, 0, stream>>>(buf1, ln2g + (size_t)l * CH,
                                             ln2b + (size_t)l * CH, hout);
    hin = hbuf;
  }
}
